// LeNet5_35708358099486
// MI455X (gfx1250) — matrix operand order repaired and checked by reading, not compiled, not run on hardware
//
#include <hip/hip_runtime.h>

// ---------------------------------------------------------------------------
// CDNA5 (gfx1250) LeNet-5 + edge-popup masking, wave32 WMMA implementation.
//   conv1: VALU direct conv fused bn+relu+pool (K=25 too small for WMMA tiles)
//   conv2: implicit-GEMM on v_wmma_f32_16x16x32_f16, per-image LDS staging,
//          im2col LDS offsets folded to compile-time constants per lane-half
//   fc1-3: v_wmma_f32_16x16x32_f16 GEMMs, K-loop fully unrolled via template
// ---------------------------------------------------------------------------

typedef _Float16 v16h __attribute__((ext_vector_type(16)));
typedef _Float16 v8h  __attribute__((ext_vector_type(8)));
typedef float    v8f  __attribute__((ext_vector_type(8)));

__device__ __forceinline__ v16h cat8(v8h lo, v8h hi) {
  return __builtin_shufflevector(lo, hi, 0, 1, 2, 3, 4, 5, 6, 7,
                                         8, 9, 10, 11, 12, 13, 14, 15);
}

__device__ __forceinline__ v8f wmma_f16(v16h a, v16h b, v8f c) {
  // (neg_a, A, neg_b, B, c_mod, C, reuse_a, reuse_b)
  return __builtin_amdgcn_wmma_f32_16x16x32_f16(false, a, false, b,
                                                (short)0, c, false, false);
}

// im2col LDS offset for flat K index k in the halo-padded [6][15][15] image.
__device__ constexpr int offc(int k) {
  return (k / 25) * 225 + ((k % 25) / 5) * 15 + (k % 5);
}

#define BN_EPS 1e-5f

// ---------------- workspace layout (bytes) ----------------
#define O_W1   ((size_t)0)                     // 150 f32 masked conv1 weights
#define O_W2P  ((size_t)1024)                  // f16 [16][160]
#define O_W3P  ((size_t)8192)                  // f16 [128][416]
#define O_W4P  ((size_t)(8192 + 106496))       // f16 [96][128]   = 114688
#define O_W5P  ((size_t)(114688 + 24576))      // f16 [16][96]    = 139264
#define O_WEND ((size_t)(139264 + 3072))       // 142336
#define O_A1   ((size_t)(1u << 20))            // f16 [8192][6][13][13]
#define O_A3   O_A1                            // f16 [8192][128] (A1 dead)
#define O_A4   (O_A1 + ((size_t)4 << 20))      // f16 [8192][96]
#define O_A2   ((size_t)(18u << 20))           // f16 [8192][416]
#define A2_BYTES ((size_t)8192 * 416 * 2)

// ---------------------------------------------------------------------------
// Top-k(0.5) mask by |score| rank (stable-sort semantics: ties break by index),
// optional f32 output (conv1) and padded-f16 pack (GEMM weights, [N][Kpad]).
// ---------------------------------------------------------------------------
__global__ __launch_bounds__(256) void mask_pack_kernel(
    const float* __restrict__ w, const float* __restrict__ s, int n, int j,
    float* __restrict__ out32, _Float16* __restrict__ out16, int K, int Kpad) {
  __shared__ float tile[256];
  const int i = blockIdx.x * 256 + threadIdx.x;
  const float my = (i < n) ? fabsf(s[i]) : 0.0f;
  int cnt = 0;
  for (int base = 0; base < n; base += 256) {
    const int l = base + threadIdx.x;
    tile[threadIdx.x] = (l < n) ? fabsf(s[l]) : 0.0f;
    __syncthreads();
    const int lim = min(256, n - base);
    for (int t = 0; t < lim; ++t) {
      const float o = tile[t];
      const int li = base + t;
      cnt += (int)((o < my) || ((o == my) && (li < i)));
    }
    __syncthreads();
  }
  if (i < n) {
    const float val = (cnt >= j) ? w[i] : 0.0f;
    if (out32) out32[i] = val;
    if (out16) out16[(i / K) * Kpad + (i % K)] = (_Float16)val;
  }
}

// ---------------------------------------------------------------------------
// conv1 (1->6ch, 5x5, pad1) + BN + ReLU + 2x2 maxpool, fully fused.
// One thread = one pooled pixel (b, py, px), all 6 channels. Output f16.
// ---------------------------------------------------------------------------
__global__ __launch_bounds__(256) void conv1_fused(
    const float* __restrict__ x, const float* __restrict__ w1m,
    const float* __restrict__ b1, const float* __restrict__ g1,
    const float* __restrict__ be1, const float* __restrict__ m1,
    const float* __restrict__ v1, _Float16* __restrict__ A1, int total) {
  const int t = blockIdx.x * 256 + threadIdx.x;
  if (t >= total) return;
  const int b = t / 169, p = t % 169, py = p / 13, px = p % 13;
  const float* xim = x + (size_t)b * 784;
  float win[6][6];
  const int r0 = 2 * py - 1, c0 = 2 * px - 1;
#pragma unroll
  for (int jj = 0; jj < 6; ++jj)
#pragma unroll
    for (int kk = 0; kk < 6; ++kk) {
      const int iy = r0 + jj, ix = c0 + kk;
      win[jj][kk] = (iy >= 0 && iy < 28 && ix >= 0 && ix < 28)
                        ? xim[iy * 28 + ix] : 0.0f;
    }
#pragma unroll
  for (int ch = 0; ch < 6; ++ch) {
    float s00 = 0.f, s01 = 0.f, s10 = 0.f, s11 = 0.f;
#pragma unroll
    for (int ky = 0; ky < 5; ++ky)
#pragma unroll
      for (int kx = 0; kx < 5; ++kx) {
        const float wt = w1m[ch * 25 + ky * 5 + kx];
        s00 = fmaf(win[ky][kx], wt, s00);
        s01 = fmaf(win[ky][kx + 1], wt, s01);
        s10 = fmaf(win[ky + 1][kx], wt, s10);
        s11 = fmaf(win[ky + 1][kx + 1], wt, s11);
      }
    const float inv = g1[ch] * rsqrtf(v1[ch] + BN_EPS);
    const float off = b1[ch] * inv + be1[ch] - m1[ch] * inv;
    const float a = fmaxf(fmaf(s00, inv, off), 0.f);
    const float bb = fmaxf(fmaf(s01, inv, off), 0.f);
    const float c = fmaxf(fmaf(s10, inv, off), 0.f);
    const float d = fmaxf(fmaf(s11, inv, off), 0.f);
    const float pooled = fmaxf(fmaxf(a, bb), fmaxf(c, d));
    A1[(((size_t)b * 6 + ch) * 13 + py) * 13 + px] = (_Float16)pooled;
  }
}

// ---------------------------------------------------------------------------
// conv2 (6->16ch, 5x5, pad1) as implicit GEMM on WMMA + BN + ReLU + pool.
// One block per image; 8 waves each own one 16-row im2col M-tile (121 rows
// padded to 128), N=16 output channels, K=150 padded to 160 (5 k-steps).
// Gather offsets are compile-time constants selected per lane-half, so the
// per-load cost is ~1 cndmask + 1 add + ds_load_u16 (keeps the WMMA pipe fed).
// ---------------------------------------------------------------------------
__global__ __launch_bounds__(256) void conv2_fused(
    const _Float16* __restrict__ A1, const _Float16* __restrict__ W2p,
    const float* __restrict__ b2, const float* __restrict__ g2,
    const float* __restrict__ be2, const float* __restrict__ m2,
    const float* __restrict__ v2, _Float16* __restrict__ A2) {
  __shared__ _Float16 simg[6 * 15 * 15];   // halo-padded input image
  __shared__ _Float16 sres[121 * 16];      // conv+bn+relu result [pix][oc]
  const int b = blockIdx.x;
  const int tid = threadIdx.x;

  for (int idx = tid; idx < 6 * 15 * 15; idx += 256) {
    const int ic = idx / 225, r = (idx % 225) / 15, c = idx % 15;
    _Float16 v = (_Float16)0.0f;
    if (r >= 1 && r <= 13 && c >= 1 && c <= 13)
      v = A1[(((size_t)b * 6 + ic) * 13 + (r - 1)) * 13 + (c - 1)];
    simg[idx] = v;
  }
  __syncthreads();

  const int wave = tid >> 5, lane = tid & 31;
  const int l15 = lane & 15;
  const bool lo = (lane < 16);
  const int bk = lo ? 0 : 8;                // f16 B-fragment K base
  const int m = wave * 16 + l15;
  const int mm = (m < 121) ? m : 0;         // pad rows: compute row 0, discard
  const int oy = mm / 11, ox = mm % 11;
  const int base = oy * 15 + ox;            // loop-invariant part of gather

  v8f acc = {};
#pragma unroll
  for (int ks = 0; ks < 5; ++ks) {
    v16h a;
#pragma unroll
    for (int h = 0; h < 8; ++h) {
      // f16 A-fragment layout (ISA 7.12.2): lanes 0-15 hold K {0..7,16..23},
      // lanes 16-31 hold K {8..15,24..31} within each 32-wide k-step.
      const int ka = ks * 32 + h;           // lo half, elems 0..7
      const int kb = ks * 32 + 8 + h;       // hi half, elems 0..7
      const int kc = ka + 16;               // lo half, elems 8..15
      const int kd = kb + 16;               // hi half, elems 8..15
      const int o0 = lo ? offc(ka < 150 ? ka : 0) : offc(kb < 150 ? kb : 0);
      const int o1 = lo ? offc(kc < 150 ? kc : 0) : offc(kd < 150 ? kd : 0);
      const _Float16 x0 = simg[base + o0];
      const _Float16 x1 = simg[base + o1];
      const bool v0 = lo ? (ka < 150) : (kb < 150);
      const bool v1 = lo ? (kc < 150) : (kd < 150);
      a[h] = v0 ? x0 : (_Float16)0.0f;
      a[8 + h] = v1 ? x1 : (_Float16)0.0f;
    }
    const v16h bf = cat8(*(const v8h*)(W2p + l15 * 160 + ks * 32 + bk),
                         *(const v8h*)(W2p + l15 * 160 + ks * 32 + bk + 16));
    acc = wmma_f16(a, bf, acc);
  }

  // BN + ReLU epilogue; column (=oc) is fixed per lane.
  const float inv = g2[l15] * rsqrtf(v2[l15] + BN_EPS);
  const float off = b2[l15] * inv + be2[l15] - m2[l15] * inv;
  const int mbase = wave * 16 + ((lane >> 4) << 3);
#pragma unroll
  for (int r = 0; r < 8; ++r) {
    const int mr = mbase + r;
    const float val = fmaxf(fmaf(acc[r], inv, off), 0.0f);
    if (mr < 121) sres[mr * 16 + l15] = (_Float16)val;
  }
  __syncthreads();

  // 2x2 maxpool 11x11 -> 5x5, flatten [oc][py][px] into FC row of width 416.
  for (int idx = tid; idx < 400; idx += 256) {
    const int oc = idx / 25, pp = idx % 25, pyy = pp / 5, pxx = pp % 5;
    const int m00 = (2 * pyy) * 11 + 2 * pxx;
    const float q0 = (float)sres[m00 * 16 + oc];
    const float q1 = (float)sres[(m00 + 1) * 16 + oc];
    const float q2 = (float)sres[(m00 + 11) * 16 + oc];
    const float q3 = (float)sres[(m00 + 12) * 16 + oc];
    A2[(size_t)b * 416 + idx] = (_Float16)fmaxf(fmaxf(q0, q1), fmaxf(q2, q3));
  }
}

// ---------------------------------------------------------------------------
// Generic FC GEMM: Y[M,Npad] = act(X[M,Kpad] * W[Npad,Kpad]^T + bias).
// One wave per 16x16 tile; fragments loaded directly from global in the
// native f16 WMMA VGPR layouts (all row strides are 32-byte multiples).
// KSTEPS is a template parameter so the k-loop fully unrolls (pipelined
// fragment loads + static wmma chain).
// ---------------------------------------------------------------------------
template <int KSTEPS>
__global__ __launch_bounds__(256) void fc_wmma(
    const _Float16* __restrict__ A, int lda, const _Float16* __restrict__ W,
    const float* __restrict__ bias, int realN, int tilesN, int numTiles,
    int relu, _Float16* __restrict__ out16, int ldo,
    float* __restrict__ out32, int on) {
  const int wave = threadIdx.x >> 5, lane = threadIdx.x & 31;
  const int tile = blockIdx.x * 8 + wave;
  if (tile >= numTiles) return;
  const int tm = tile / tilesN, tn = tile % tilesN;
  const int l15 = lane & 15;
  const int kb0 = (lane < 16) ? 0 : 8;
  const int bk = (lane < 16) ? 0 : 16;
  const _Float16* arow = A + (size_t)(tm * 16 + l15) * lda;
  const _Float16* wrow = W + (size_t)(tn * 16 + l15) * lda;

  v8f acc = {};
#pragma unroll
  for (int ks = 0; ks < KSTEPS; ++ks) {
    const _Float16* ap = arow + ks * 32;
    const v16h a = cat8(*(const v8h*)(ap + kb0), *(const v8h*)(ap + kb0 + 16));
    const v16h bf = cat8(*(const v8h*)(wrow + ks * 32 + kb0), *(const v8h*)(wrow + ks * 32 + kb0 + 16));
    acc = wmma_f16(a, bf, acc);
  }

  const int n = tn * 16 + l15;
  const float bv = (n < realN) ? bias[n] : 0.0f;
  const int mbase = tm * 16 + ((lane >> 4) << 3);
#pragma unroll
  for (int r = 0; r < 8; ++r) {
    float val = acc[r] + bv;
    if (relu) val = fmaxf(val, 0.0f);
    const int mr = mbase + r;
    if (out16) {
      out16[(size_t)mr * ldo + n] = (_Float16)val;
    } else if (n < on) {
      out32[(size_t)mr * on + n] = val;
    }
  }
}

// ---------------------------------------------------------------------------
extern "C" void kernel_launch(void* const* d_in, const int* in_sizes, int n_in,
                              void* d_out, int out_size, void* d_ws,
                              size_t ws_size, hipStream_t stream) {
  (void)in_sizes; (void)n_in; (void)out_size; (void)ws_size;
  const float* x   = (const float*)d_in[0];
  const float* w1  = (const float*)d_in[1];
  const float* b1  = (const float*)d_in[2];
  const float* s1  = (const float*)d_in[3];
  const float* g1  = (const float*)d_in[4];
  const float* be1 = (const float*)d_in[5];
  const float* m1  = (const float*)d_in[6];
  const float* v1  = (const float*)d_in[7];
  const float* w2  = (const float*)d_in[8];
  const float* b2  = (const float*)d_in[9];
  const float* s2  = (const float*)d_in[10];
  const float* g2  = (const float*)d_in[11];
  const float* be2 = (const float*)d_in[12];
  const float* m2  = (const float*)d_in[13];
  const float* v2  = (const float*)d_in[14];
  const float* w3  = (const float*)d_in[15];
  const float* b3  = (const float*)d_in[16];
  const float* s3  = (const float*)d_in[17];
  const float* w4  = (const float*)d_in[18];
  const float* b4  = (const float*)d_in[19];
  const float* s4  = (const float*)d_in[20];
  const float* w5  = (const float*)d_in[21];
  const float* b5  = (const float*)d_in[22];
  const float* s5  = (const float*)d_in[23];

  char* ws = (char*)d_ws;
  float*    w1m = (float*)(ws + O_W1);
  _Float16* w2p = (_Float16*)(ws + O_W2P);
  _Float16* w3p = (_Float16*)(ws + O_W3P);
  _Float16* w4p = (_Float16*)(ws + O_W4P);
  _Float16* w5p = (_Float16*)(ws + O_W5P);
  _Float16* A1  = (_Float16*)(ws + O_A1);
  _Float16* A2  = (_Float16*)(ws + O_A2);
  _Float16* A3  = (_Float16*)(ws + O_A3);
  _Float16* A4  = (_Float16*)(ws + O_A4);
  float* out = (float*)d_out;

  // Zero the padded weight arrays + padded FC input (graph-capture safe).
  hipMemsetAsync(ws, 0, O_WEND, stream);
  hipMemsetAsync(ws + O_A2, 0, A2_BYTES, stream);

  // Masked / packed weights.
  mask_pack_kernel<<<1, 256, 0, stream>>>(w1, s1, 150, 75, w1m, nullptr, 25, 25);
  mask_pack_kernel<<<10, 256, 0, stream>>>(w2, s2, 2400, 1200, nullptr, w2p, 150, 160);
  mask_pack_kernel<<<188, 256, 0, stream>>>(w3, s3, 48000, 24000, nullptr, w3p, 400, 416);
  mask_pack_kernel<<<40, 256, 0, stream>>>(w4, s4, 10080, 5040, nullptr, w4p, 120, 128);
  mask_pack_kernel<<<4, 256, 0, stream>>>(w5, s5, 840, 420, nullptr, w5p, 84, 96);

  // conv1 + bn + relu + pool (VALU, fused).
  conv1_fused<<<5408, 256, 0, stream>>>(x, w1m, b1, g1, be1, m1, v1, A1,
                                        8192 * 169);
  // conv2 + bn + relu + pool (WMMA implicit GEMM, one block per image).
  conv2_fused<<<8192, 256, 0, stream>>>(A1, w2p, b2, g2, be2, m2, v2, A2);

  // FC layers on WMMA. M=8192 -> 512 M-tiles.
  fc_wmma<13><<<512, 256, 0, stream>>>(A2, 416, w3p, b3, 120, 8, 4096, 1,
                                       A3, 128, nullptr, 0);
  fc_wmma<4><<<384, 256, 0, stream>>>(A3, 128, w4p, b4, 84, 6, 3072, 1,
                                      A4, 96, nullptr, 0);
  fc_wmma<3><<<64, 256, 0, stream>>>(A4, 96, w5p, b5, 10, 1, 512, 0,
                                     nullptr, 0, out, 10);
}
